// Multihead_attention_83013127897630
// MI455X (gfx1250) — matrix operand order repaired and checked by reading, not compiled, not run on hardware
//
#include <hip/hip_runtime.h>

#define B_  8
#define LQ_ 1024
#define LK_ 1024
#define E_  512
#define H_  8
#define D_  64

#define KROWS 64          // keys per staged block
#define LDS_PITCH 72      // padded row pitch (bf16 elems): 144B = 36 dwords, kills bank conflicts

typedef __attribute__((ext_vector_type(16))) __bf16 v16bf;
typedef __attribute__((ext_vector_type(8)))  __bf16 v8bf;
typedef __attribute__((ext_vector_type(4)))  __bf16 v4bf;
typedef __attribute__((ext_vector_type(8)))  float  v8f;
typedef __attribute__((ext_vector_type(4)))  float  v4f;
typedef __attribute__((ext_vector_type(4)))  int    v4i;

static __device__ __forceinline__ v8f wmma_bf16(v16bf a, v16bf b, v8f c) {
  // D = A(16x32 bf16) x B(32x16 bf16) + C(16x16 f32)
  return __builtin_amdgcn_wmma_f32_16x16x32_bf16(false, a, false, b, (short)0, c,
                                                 false, false);
}

static __device__ __forceinline__ v8f v8f_zero() {
  v8f z;
#pragma unroll
  for (int r = 0; r < 8; ++r) z[r] = 0.0f;
  return z;
}

// ---- async 16B global -> LDS copy (ASYNCcnt-tracked on gfx1250) ------------
static __device__ __forceinline__ void async_cp16(const __bf16* g, __bf16* l) {
#if __has_builtin(__builtin_amdgcn_global_load_async_to_lds_b128)
  __builtin_amdgcn_global_load_async_to_lds_b128(
      (__attribute__((address_space(1))) v4i*)g,
      (__attribute__((address_space(3))) v4i*)l, 0, 0);
#else
  *(v8bf*)l = *(const v8bf*)g;  // fallback: load + ds_store
#endif
}

template <int N>
static __device__ __forceinline__ void wait_async() {
#if __has_builtin(__builtin_amdgcn_s_wait_asynccnt)
  __builtin_amdgcn_s_wait_asynccnt(N);
#else
  asm volatile("s_wait_asynccnt %0" ::"i"(N) : "memory");
#endif
}

// ---------------------------------------------------------------- fp32->bf16
__global__ __launch_bounds__(256) void cvt_f32_to_bf16(
    const float* __restrict__ src, __bf16* __restrict__ dst, int n4) {
  int i = blockIdx.x * blockDim.x + threadIdx.x;
  if (i >= n4) return;
  v4f x = *(const v4f*)(src + (size_t)i * 4);
  v4bf y;
#pragma unroll
  for (int j = 0; j < 4; ++j) y[j] = (__bf16)x[j];
  *(v4bf*)(dst + (size_t)i * 4) = y;
}

// ------------------------------------------------- mask = sin(|sum(row)|)
__global__ __launch_bounds__(256) void mask_kernel(
    const float* __restrict__ X, float* __restrict__ mask, int rows) {
  int row = blockIdx.x * (blockDim.x >> 5) + (threadIdx.x >> 5);
  if (row >= rows) return;
  int lane = threadIdx.x & 31;
  const float* p = X + (size_t)row * E_;
  float s = 0.0f;
#pragma unroll
  for (int i = 0; i < E_ / 32; ++i) s += p[lane + 32 * i];
#pragma unroll
  for (int ofs = 16; ofs >= 1; ofs >>= 1) s += __shfl_xor(s, ofs, 32);
  if (lane == 0) mask[row] = sinf(fabsf(s));
}

// -------------------------- C = relu(X(bf16) @ W(bf16)^T + bias), 3 store modes
// mode 0: fp32 row-major (Q, kept fp32 for residual)
// mode 1: bf16 row-major (K)
// mode 2: bf16 transposed per (b, e): vT[(b*E + e)*LK + key]   (V)
__global__ __launch_bounds__(256) void proj_kernel(
    const __bf16* __restrict__ X, const __bf16* __restrict__ W,
    const float* __restrict__ bias, float* __restrict__ outF,
    __bf16* __restrict__ outB, __bf16* __restrict__ outVT, int mode) {
  const int lane = threadIdx.x & 31;
  const int nn = lane & 15, hi = lane >> 4;
  const int tile = blockIdx.x * 8 + (threadIdx.x >> 5);
  const int m0 = (tile >> 5) * 16;   // 512 row-tiles
  const int n0 = (tile & 31) * 16;   // 32 col-tiles

  const __bf16* xrow = X + (size_t)(m0 + nn) * E_;  // A: lane holds row m=nn
  const __bf16* wrow = W + (size_t)(n0 + nn) * E_;  // B: lane holds col n=nn

  v8f c = v8f_zero();
#pragma unroll 4
  for (int k0 = 0; k0 < E_; k0 += 32) {
    v16bf a;
    // A 16x32 bf16, contiguous k order used here (B matches): elems 0..15 -> K = hi*16+j
    ((v8bf*)&a)[0] = *(const v8bf*)(xrow + k0 + hi * 16);
    ((v8bf*)&a)[1] = *(const v8bf*)(xrow + k0 + hi * 16 + 8);
    // B 32x16 bf16, same contiguous k order as A: lane group hi covers K = hi*16 .. hi*16+15
    v16bf bm = *(const v16bf*)(wrow + k0 + hi * 16);
    c = wmma_bf16(a, bm, c);
  }
  float bv = bias[n0 + nn];
#pragma unroll
  for (int r = 0; r < 8; ++r) {
    int mrow = m0 + r + 8 * hi;  // C layout: VGPR r -> row r + 8*hi, col nn
    float v = fmaxf(c[r] + bv, 0.0f);
    if (mode == 0) {
      outF[(size_t)mrow * E_ + n0 + nn] = v;
    } else if (mode == 1) {
      outB[(size_t)mrow * E_ + n0 + nn] = (__bf16)v;
    } else {
      int bb = mrow >> 10, key = mrow & (LK_ - 1);
      outVT[((size_t)bb * E_ + n0 + nn) * LK_ + key] = (__bf16)v;
    }
  }
}

// ---- cooperative async stage of one 64-key K block + V block into LDS ------
static __device__ __forceinline__ void stage_kv(
    const __bf16* __restrict__ kbf, const __bf16* __restrict__ vT,
    __bf16* lk, __bf16* lv, int b, int h, int kc, int tid) {
#pragma unroll
  for (int i = 0; i < 4; ++i) {
    int c = i * 128 + tid;            // 512 x 16B chunks per matrix
    int row = c >> 3;                 // 64 rows
    int col = (c & 7) * 8;            // 8 x 16B per 128B row
    const __bf16* gk =
        kbf + ((size_t)b * LK_ + kc + row) * E_ + h * D_ + col;
    async_cp16(gk, lk + row * LDS_PITCH + col);
    const __bf16* gv =
        vT + ((size_t)b * E_ + h * D_ + row) * LK_ + kc + col;
    async_cp16(gv, lv + row * LDS_PITCH + col);
  }
}

// ------------------------------------------------ fused flash attention
__global__ __launch_bounds__(128) void attn_kernel(
    const float* __restrict__ qf, const __bf16* __restrict__ kbf,
    const __bf16* __restrict__ vT, const float* __restrict__ key_mask,
    const float* __restrict__ query_mask, float* __restrict__ out) {
  const int h = blockIdx.z, b = blockIdx.y;
  const int wave = threadIdx.x >> 5;
  const int lane = threadIdx.x & 31;
  const int nn = lane & 15, hi = lane >> 4;
  const int qrow0 = blockIdx.x * 64 + wave * 16;  // 16-row strip per wave

  // double-buffered K/V staging + per-wave P relayout buffer (45 KB total)
  __shared__ __align__(16) __bf16 lds_k[2][KROWS * LDS_PITCH];
  __shared__ __align__(16) __bf16 lds_v[2][KROWS * LDS_PITCH];
  __shared__ __align__(16) __bf16 lds_p[4][16 * LDS_PITCH];

  const float SCALE = 0.04419417382415922f;  // 1/sqrt(E)
  const float NEGV = -4294967295.0f;         // -2^32 + 1

  // ---- load Q strip as two A-tiles (fp32 -> bf16, scale folded in)
  v16bf aq[2];
  {
    const float* pr = qf + ((size_t)b * LQ_ + qrow0 + nn) * E_ + h * D_;
#pragma unroll
    for (int ks = 0; ks < 2; ++ks) {
      const float* p = pr + ks * 32 + hi * 16;
#pragma unroll
      for (int j = 0; j < 8; ++j) aq[ks][j] = (__bf16)(p[j] * SCALE);
#pragma unroll
      for (int j = 0; j < 8; ++j) aq[ks][8 + j] = (__bf16)(p[8 + j] * SCALE);
    }
  }

  v8f o[4];
#pragma unroll
  for (int t = 0; t < 4; ++t) o[t] = v8f_zero();
  float mrow[8], lrow[8];
#pragma unroll
  for (int r = 0; r < 8; ++r) { mrow[r] = -3.0e38f; lrow[r] = 0.0f; }

  // prologue: stage block 0
  stage_kv(kbf, vT, lds_k[0], lds_v[0], b, h, 0, threadIdx.x);

  for (int kb = 0; kb < LK_ / KROWS; ++kb) {
    const int cur = kb & 1;
    const int kc = kb * KROWS;

    // issue next stage into the other buffer, retire only the previous one
    if (kb + 1 < LK_ / KROWS) {
      stage_kv(kbf, vT, lds_k[cur ^ 1], lds_v[cur ^ 1], b, h, kc + KROWS,
               threadIdx.x);
      wait_async<8>();   // 8 new in flight; <=8 means stage(kb) retired
    } else {
      wait_async<0>();
    }
    __syncthreads();     // staged K/V visible to all waves

    // ---- S = Qs · K^T from LDS (B tile (d,key): lane = key col)
    v8f s[4];
#pragma unroll
    for (int t = 0; t < 4; ++t) s[t] = v8f_zero();
#pragma unroll
    for (int ks = 0; ks < 2; ++ks) {
#pragma unroll
      for (int t = 0; t < 4; ++t) {
        v16bf bk = *(const v16bf*)&lds_k[cur][(t * 16 + nn) * LDS_PITCH +
                                              ks * 32 + hi * 16];
        s[t] = wmma_bf16(aq[ks], bk, s[t]);
      }
    }

    // ---- key mask (column-wise): mask<=0 -> NEG
#pragma unroll
    for (int t = 0; t < 4; ++t) {
      float km = key_mask[(size_t)b * LK_ + kc + t * 16 + nn];
      if (km <= 0.0f) {
#pragma unroll
        for (int r = 0; r < 8; ++r) s[t][r] = NEGV;
      }
    }

    // ---- online softmax; each row lives in one 16-lane half
#pragma unroll
    for (int r = 0; r < 8; ++r) {
      float mx = fmaxf(fmaxf(s[0][r], s[1][r]), fmaxf(s[2][r], s[3][r]));
      mx = fmaxf(mx, __shfl_xor(mx, 1, 32));
      mx = fmaxf(mx, __shfl_xor(mx, 2, 32));
      mx = fmaxf(mx, __shfl_xor(mx, 4, 32));
      mx = fmaxf(mx, __shfl_xor(mx, 8, 32));
      float mnew = fmaxf(mrow[r], mx);
      float alpha = __expf(mrow[r] - mnew);
      mrow[r] = mnew;
      float rs = 0.0f;
#pragma unroll
      for (int t = 0; t < 4; ++t) {
        float p = __expf(s[t][r] - mnew);
        s[t][r] = p;
        rs += p;
      }
      rs += __shfl_xor(rs, 1, 32);
      rs += __shfl_xor(rs, 2, 32);
      rs += __shfl_xor(rs, 4, 32);
      rs += __shfl_xor(rs, 8, 32);
      lrow[r] = lrow[r] * alpha + rs;
#pragma unroll
      for (int t = 0; t < 4; ++t) o[t][r] *= alpha;
    }

    // ---- relayout P (C layout -> A layout) via per-wave padded LDS tile
#pragma unroll
    for (int t = 0; t < 4; ++t)
#pragma unroll
      for (int r = 0; r < 8; ++r)
        lds_p[wave][(r + 8 * hi) * LDS_PITCH + t * 16 + nn] = (__bf16)s[t][r];
    asm volatile("s_wait_dscnt 0x0" ::: "memory");  // wave-internal RAW on LDS

    // ---- O += P · V from LDS
#pragma unroll
    for (int ks = 0; ks < 2; ++ks) {
      v16bf ap;
      ((v8bf*)&ap)[0] =
          *(const v8bf*)&lds_p[wave][nn * LDS_PITCH + ks * 32 + hi * 16];
      ((v8bf*)&ap)[1] =
          *(const v8bf*)&lds_p[wave][nn * LDS_PITCH + ks * 32 + hi * 16 + 8];
#pragma unroll
      for (int t = 0; t < 4; ++t) {
        v16bf bv = *(const v16bf*)&lds_v[cur][(t * 16 + nn) * LDS_PITCH +
                                              ks * 32 + hi * 16];
        o[t] = wmma_bf16(ap, bv, o[t]);
      }
    }
    __syncthreads();  // everyone done reading buf[cur] before it is restaged
  }

  // ---- epilogue: 1/l, query mask, fp32 Q residual
#pragma unroll
  for (int r = 0; r < 8; ++r) {
    int row = qrow0 + r + 8 * hi;
    float invl = 1.0f / lrow[r];
    float qm = query_mask[(size_t)b * LQ_ + row];
#pragma unroll
    for (int t = 0; t < 4; ++t) {
      size_t oidx = ((size_t)b * LQ_ + row) * E_ + h * D_ + t * 16 + nn;
      out[oidx] = o[t][r] * invl * qm + qf[oidx];
    }
  }
}

// ---------------------------------------------------------------- host
extern "C" void kernel_launch(void* const* d_in, const int* in_sizes, int n_in,
                              void* d_out, int out_size, void* d_ws,
                              size_t ws_size, hipStream_t stream) {
  (void)in_sizes; (void)n_in; (void)out_size; (void)ws_size;
  const float* x1 = (const float*)d_in[0];
  const float* x2 = (const float*)d_in[1];
  const float* Wq = (const float*)d_in[2];
  const float* bq = (const float*)d_in[3];
  const float* Wk = (const float*)d_in[4];
  const float* bk = (const float*)d_in[5];
  const float* Wv = (const float*)d_in[6];
  const float* bv = (const float*)d_in[7];
  float* out = (float*)d_out;
  char* ws = (char*)d_ws;

  // workspace carve (total ~49.6 MB)
  float*  q_f32 = (float*)(ws + 0);            // 16,777,216 B
  __bf16* x1b   = (__bf16*)(ws + 16777216);    //  8,388,608 B
  __bf16* x2b   = (__bf16*)(ws + 25165824);    //  8,388,608 B
  __bf16* wqb   = (__bf16*)(ws + 33554432);    //    524,288 B
  __bf16* wkb   = (__bf16*)(ws + 34078720);    //    524,288 B
  __bf16* wvb   = (__bf16*)(ws + 34603008);    //    524,288 B
  __bf16* kbf   = (__bf16*)(ws + 35127296);    //  8,388,608 B
  __bf16* vT    = (__bf16*)(ws + 43515904);    //  8,388,608 B
  float*  km    = (float*)(ws + 51904512);     //     32,768 B
  float*  qm    = (float*)(ws + 51937280);     //     32,768 B

  // 1) bf16 conversions
  cvt_f32_to_bf16<<<4096, 256, 0, stream>>>(x1, x1b, (B_ * LQ_ * E_) / 4);
  cvt_f32_to_bf16<<<4096, 256, 0, stream>>>(x2, x2b, (B_ * LK_ * E_) / 4);
  cvt_f32_to_bf16<<<256, 256, 0, stream>>>(Wq, wqb, (E_ * E_) / 4);
  cvt_f32_to_bf16<<<256, 256, 0, stream>>>(Wk, wkb, (E_ * E_) / 4);
  cvt_f32_to_bf16<<<256, 256, 0, stream>>>(Wv, wvb, (E_ * E_) / 4);

  // 2) sin masks from raw inputs
  mask_kernel<<<1024, 256, 0, stream>>>(x1, qm, B_ * LQ_);
  mask_kernel<<<1024, 256, 0, stream>>>(x2, km, B_ * LK_);

  // 3) relu projections (WMMA GEMMs)
  proj_kernel<<<2048, 256, 0, stream>>>(x1b, wqb, bq, q_f32, nullptr, nullptr, 0);
  proj_kernel<<<2048, 256, 0, stream>>>(x2b, wkb, bk, nullptr, kbf, nullptr, 1);
  proj_kernel<<<2048, 256, 0, stream>>>(x2b, wvb, bv, nullptr, nullptr, vT, 2);

  // 4) fused flash attention + masks + residual (async-staged K/V)
  attn_kernel<<<dim3(LQ_ / 64, B_, H_), 128, 0, stream>>>(q_f32, kbf, vT, km,
                                                          qm, out);
}
